// GCN_SP_three_before_86887188398705
// MI455X (gfx1250) — matrix operand order repaired and checked by reading, not compiled, not run on hardware
//
#include <hip/hip_runtime.h>

// GCN inference for MI455X (gfx1250, wave32, WMMA, async-to-LDS).
//   - GEMMs: bf16 V_WMMA_F32_16X16X32_BF16, 128x64 tile / 256-thread block (8 waves),
//     double-buffered LDS fed by GLOBAL_LOAD_ASYNC_TO_LDS_B128 (ASYNCcnt).
//   - All GEMM A-operands pre-converted to bf16 (halves HBM traffic; staging = pure copy).
//   - SpMM: edge-parallel f32 global atomics (targets are L2-resident: <=102MB vs 192MB L2).

typedef __attribute__((ext_vector_type(16))) __bf16 v16bf;
typedef __attribute__((ext_vector_type(8)))  __bf16 v8bf;
typedef __attribute__((ext_vector_type(8)))  float  v8f;
typedef int b128i __attribute__((vector_size(16)));   // matches async-to-LDS builtin pointee

#define NFEAT  512
#define NHID1  256
#define NHID2  128
#define NCLASS 40
#define LDT    40   // LDS row stride in bf16 elems (32 + 8 pad; 80B rows keep 16B alignment)

#if __has_builtin(__builtin_amdgcn_global_load_async_to_lds_b128)
#define USE_ASYNC_LDS 1
#else
#define USE_ASYNC_LDS 0
#endif

__device__ __forceinline__ void copy16_to_lds(__bf16* ldst, const __bf16* gsrc) {
#if USE_ASYNC_LDS
  __builtin_amdgcn_global_load_async_to_lds_b128(
      (__attribute__((address_space(1))) b128i*)(unsigned long long)(__UINTPTR_TYPE__)gsrc,
      (__attribute__((address_space(3))) b128i*)(unsigned)(__UINTPTR_TYPE__)ldst,
      0, 0);
#else
  *(v8bf*)ldst = *(const v8bf*)gsrc;
#endif
}

__device__ __forceinline__ void wait_async_all() {
#if USE_ASYNC_LDS
#if __has_builtin(__builtin_amdgcn_s_wait_asynccnt)
  __builtin_amdgcn_s_wait_asynccnt(0);
#else
  asm volatile("s_wait_asynccnt 0x0" ::: "memory");
#endif
#endif
}

// ---------------- f32 -> bf16 bulk convert (4 elems/thread) ----------------
__global__ __launch_bounds__(256)
void k_f32x4_to_bf16(const float* __restrict__ s, __bf16* __restrict__ d, long long n4) {
  long long t = (long long)blockIdx.x * blockDim.x + threadIdx.x;
  if (t >= n4) return;
  float4 v = ((const float4*)s)[t];
  __bf16* p = d + t * 4;
  p[0] = (__bf16)v.x; p[1] = (__bf16)v.y; p[2] = (__bf16)v.z; p[3] = (__bf16)v.w;
}

// ---------------- weight transpose + convert: Wt[n*K+k] = W[k*NC+n] ----------------
__global__ void k_convert_wt(const float* __restrict__ W, __bf16* __restrict__ Wt,
                             int K, int NC) {
  int tid = blockIdx.x * blockDim.x + threadIdx.x;
  if (tid >= K * NC) return;
  int n = tid / K, k = tid - n * K;
  Wt[(size_t)n * K + k] = (__bf16)W[(size_t)k * NC + n];
}

// ---------------- tiled bf16 WMMA GEMM: C[M,NC] = A[M,K] @ W[K,NC] (+bias) ----------------
// A: bf16 row-major, padded to a multiple of 128 rows (pad rows zeroed).
// Bt: bf16 W pre-transposed to [NCpad,K] row-major (pad rows zeroed) -> contiguous-K fragments.
__global__ __launch_bounds__(256)
void k_gemm_bf16(const __bf16* __restrict__ A, const __bf16* __restrict__ Bt,
                 float* __restrict__ C, int Mrows, int K, int NC,
                 const float* __restrict__ bias) {
  __shared__ __align__(16) __bf16 ldsA[2][128 * LDT];
  __shared__ __align__(16) __bf16 ldsB[2][64 * LDT];
  const int tid  = threadIdx.x;
  const int lane = tid & 31;
  const int wave = tid >> 5;
  const int wm   = wave & 3;    // 4 waves along M, 32 rows each
  const int wn   = wave >> 2;   // 2 waves along N, 32 cols each
  const int half = lane >> 4;
  const int r    = lane & 15;
  const int mb   = blockIdx.x * 128;
  const int nb   = blockIdx.y * 64;

  v8f acc[2][2];
  #pragma unroll
  for (int mi = 0; mi < 2; ++mi)
    #pragma unroll
    for (int ni = 0; ni < 2; ++ni)
      #pragma unroll
      for (int j = 0; j < 8; ++j) acc[mi][ni][j] = 0.0f;

  // stage one 128x32 A tile + 64x32 B tile (16B chunks, no guards: buffers padded)
  auto stage = [&](int buf, int kb) {
    #pragma unroll
    for (int i = 0; i < 2; ++i) {
      int c = tid + i * 256;            // 512 chunks cover 128x32 bf16
      int row = c >> 2, ch = (c & 3) * 8;
      copy16_to_lds(&ldsA[buf][row * LDT + ch],
                    A + (size_t)(mb + row) * K + kb + ch);
    }
    {
      int n = tid >> 2, ch = (tid & 3) * 8;  // 256 chunks cover 64x32 bf16
      copy16_to_lds(&ldsB[buf][n * LDT + ch],
                    Bt + (size_t)(nb + n) * K + kb + ch);
    }
  };

  stage(0, 0);
  wait_async_all();
  __syncthreads();

  const int nk = K >> 5;
  for (int t = 0; t < nk; ++t) {
    const int cur = t & 1;
    if (t + 1 < nk) stage(cur ^ 1, (t + 1) << 5);   // prefetch next tile into other buffer

    // A fragment (ISA 16-bit A 16x32): elem j<8 -> K=half*8+j ; j>=8 -> K=16+half*8+(j-8)
    v16bf af[2];
    #pragma unroll
    for (int mi = 0; mi < 2; ++mi) {
      const __bf16* base = &ldsA[cur][(wm * 32 + mi * 16 + r) * LDT];
      v8bf lo = *(const v8bf*)(base + half * 8);
      v8bf hi = *(const v8bf*)(base + 16 + half * 8);
      #pragma unroll
      for (int j = 0; j < 8; ++j) { af[mi][j] = lo[j]; af[mi][8 + j] = hi[j]; }
    }
    // B fragment (ISA 16-bit B 32x16): lane = column, elem j<8 -> K=half*8+j ; j>=8 -> K=16+half*8+(j-8)
    v16bf bfr[2];
    #pragma unroll
    for (int ni = 0; ni < 2; ++ni) {
      const __bf16* base = &ldsB[cur][(wn * 32 + ni * 16 + r) * LDT + half * 8];
      v8bf lo = *(const v8bf*)(base);
      v8bf hi = *(const v8bf*)(base + 16);
      #pragma unroll
      for (int j = 0; j < 8; ++j) { bfr[ni][j] = lo[j]; bfr[ni][8 + j] = hi[j]; }
    }

    #pragma unroll
    for (int mi = 0; mi < 2; ++mi)
      #pragma unroll
      for (int ni = 0; ni < 2; ++ni)
        acc[mi][ni] = __builtin_amdgcn_wmma_f32_16x16x32_bf16(
            false, af[mi], false, bfr[ni], (short)0, acc[mi][ni], false, false);

    wait_async_all();   // our async loads into the other buffer are done
    __syncthreads();    // everyone finished reading cur / loads visible
  }

  // store: C elem i -> M = i + 8*half, N = r (half-wave writes 64B contiguous)
  #pragma unroll
  for (int mi = 0; mi < 2; ++mi) {
    #pragma unroll
    for (int ni = 0; ni < 2; ++ni) {
      int colc = nb + wn * 32 + ni * 16 + r;
      if (colc >= NC) continue;
      float badd = bias ? bias[colc] : 0.0f;
      #pragma unroll
      for (int i = 0; i < 8; ++i) {
        int rowc = mb + wm * 32 + mi * 16 + half * 8 + i;
        if (rowc < Mrows)
          C[(size_t)rowc * NC + colc] = acc[mi][ni][i] + badd;
      }
    }
  }
}

// ---------------- edge-parallel SpMM scatter: out[dst] += w * sup[src] ----------------
template<int D, int D4>
__global__ __launch_bounds__(256)
void k_spmm_atomic(const int* __restrict__ ei, const float* __restrict__ ew,
                   const float* __restrict__ sup, float* __restrict__ out, int E) {
  long long tid = (long long)blockIdx.x * blockDim.x + threadIdx.x;
  if (tid >= (long long)E * D4) return;
  int e = (int)(tid / D4);
  int c = (int)(tid % D4) * 4;
  int src = ei[e];
  int dst = ei[(size_t)E + e];
  float w = ew[e];
  float4 v = *(const float4*)(sup + (size_t)src * D + c);
  float* o = out + (size_t)dst * D + c;
  atomicAdd(o + 0, v.x * w);
  atomicAdd(o + 1, v.y * w);
  atomicAdd(o + 2, v.z * w);
  atomicAdd(o + 3, v.w * w);
}

// ---------------- bias + relu, f32 agg -> bf16 activations (next layer's A) ----------------
template<int D>
__global__ __launch_bounds__(256)
void k_bias_relu_bf16(const float* __restrict__ x, const float* __restrict__ b,
                      __bf16* __restrict__ y, long long n) {
  long long tid = (long long)blockIdx.x * blockDim.x + threadIdx.x;
  if (tid >= n) return;
  int col = (int)(tid % D);
  y[tid] = (__bf16)fmaxf(x[tid] + b[col], 0.0f);
}

// ---------------- bias + log_softmax over 40 classes; one wave32 per row ----------------
__global__ __launch_bounds__(256)
void k_logsoftmax40(const float* __restrict__ agg, const float* __restrict__ b,
                    float* __restrict__ out, int Nrows) {
  int wave = threadIdx.x >> 5, lane = threadIdx.x & 31;
  int row = blockIdx.x * 8 + wave;
  if (row >= Nrows) return;
  const float* x = agg + (size_t)row * NCLASS;
  float v0 = x[lane] + b[lane];
  float v1 = (lane < 8) ? (x[32 + lane] + b[32 + lane]) : -3.402823e38f;
  float m = fmaxf(v0, v1);
  #pragma unroll
  for (int o = 16; o > 0; o >>= 1) m = fmaxf(m, __shfl_xor(m, o, 32));
  float s = __expf(v0 - m) + ((lane < 8) ? __expf(v1 - m) : 0.0f);
  #pragma unroll
  for (int o = 16; o > 0; o >>= 1) s += __shfl_xor(s, o, 32);
  float ls = __logf(s);
  out[(size_t)row * NCLASS + lane] = v0 - m - ls;
  if (lane < 8) out[(size_t)row * NCLASS + 32 + lane] = v1 - m - ls;
}

extern "C" void kernel_launch(void* const* d_in, const int* in_sizes, int n_in,
                              void* d_out, int out_size, void* d_ws, size_t ws_size,
                              hipStream_t stream) {
  const float* x  = (const float*)d_in[0];
  const int*   ei = (const int*)  d_in[1];   // [2,E] : row0 = src, row1 = dst
  const float* ew = (const float*)d_in[2];
  const float* W1 = (const float*)d_in[3];
  const float* b1 = (const float*)d_in[4];
  const float* W2 = (const float*)d_in[5];
  const float* b2 = (const float*)d_in[6];
  const float* W3 = (const float*)d_in[7];
  const float* b3 = (const float*)d_in[8];
  const float* We = (const float*)d_in[9];
  const float* be = (const float*)d_in[10];
  float* out = (float*)d_out;

  const int N    = in_sizes[0] / NFEAT;
  const int E    = in_sizes[2];
  const int Mpad = ((N + 127) / 128) * 128;

  char* ws = (char*)d_ws;
  size_t off = 0;
  float*  s_buf = (float*)(ws + off);  off += (size_t)N * NHID1 * 4;      // support (reused)
  float*  agg   = (float*)(ws + off);  off += (size_t)N * NHID1 * 4;      // agg1/agg2/agg3
  __bf16* abf   = (__bf16*)(ws + off); off += (size_t)Mpad * NFEAT * 2;   // bf16 x (layer-1 A)
  __bf16* wt1   = (__bf16*)(ws + off); off += (size_t)NFEAT * NHID1 * 2;
  __bf16* wt2   = (__bf16*)(ws + off); off += (size_t)NHID1 * NHID2 * 2;
  __bf16* wt3   = (__bf16*)(ws + off); off += (size_t)NHID2 * 64 * 2;     // padded to 64 rows
  __bf16* wte   = (__bf16*)(ws + off); off += (size_t)NHID1 * NHID2 * 2;
  // abf is dead after GEMM1 -> alias bf16 h and t into its footprint
  __bf16* hbf = abf;                                        // [Mpad, 256] bf16
  __bf16* gbf = abf + (size_t)Mpad * NHID1;                 // [Mpad, 128] bf16 (fits: 256+128 <= 512)

  float* out1 = out;                        // [N, 40]  log_softmax
  float* out2 = out + (size_t)N * NCLASS;   // [N, 128]

  dim3 blk(256);

  // ---- prep: weights (tiny) + x -> bf16 with zeroed row padding ----
  k_convert_wt<<<(NFEAT * NHID1 + 255) / 256, blk, 0, stream>>>(W1, wt1, NFEAT, NHID1);
  k_convert_wt<<<(NHID1 * NHID2 + 255) / 256, blk, 0, stream>>>(W2, wt2, NHID1, NHID2);
  k_convert_wt<<<(NHID2 * NCLASS + 255) / 256, blk, 0, stream>>>(W3, wt3, NHID2, NCLASS);
  (void)hipMemsetAsync(wt3 + (size_t)NCLASS * NHID2, 0, (size_t)(64 - NCLASS) * NHID2 * 2, stream);
  k_convert_wt<<<(NHID1 * NHID2 + 255) / 256, blk, 0, stream>>>(We, wte, NHID1, NHID2);
  {
    long long n4 = (long long)N * NFEAT / 4;
    k_f32x4_to_bf16<<<(unsigned)((n4 + 255) / 256), blk, 0, stream>>>(x, abf, n4);
    (void)hipMemsetAsync(abf + (size_t)N * NFEAT, 0, (size_t)(Mpad - N) * NFEAT * 2, stream);
  }

  // ---- layer 1: s = x @ W1 ; h = relu(spmm(s) + b1) (bf16) ----
  {
    dim3 grid(Mpad / 128, NHID1 / 64);
    k_gemm_bf16<<<grid, blk, 0, stream>>>(abf, wt1, s_buf, N, NFEAT, NHID1, nullptr);
  }
  (void)hipMemsetAsync(agg, 0, (size_t)N * NHID1 * 4, stream);
  {
    long long total = (long long)E * (NHID1 / 4);
    k_spmm_atomic<NHID1, NHID1 / 4><<<(unsigned)((total + 255) / 256), blk, 0, stream>>>(ei, ew, s_buf, agg, E);
  }
  k_bias_relu_bf16<NHID1><<<(unsigned)(((long long)N * NHID1 + 255) / 256), blk, 0, stream>>>(
      agg, b1, hbf, (long long)N * NHID1);
  (void)hipMemsetAsync(hbf + (size_t)N * NHID1, 0, (size_t)(Mpad - N) * NHID1 * 2, stream);

  // ---- out2 = h @ We + be (straight into d_out, f32) ----
  {
    dim3 grid(Mpad / 128, NHID2 / 64);
    k_gemm_bf16<<<grid, blk, 0, stream>>>(hbf, wte, out2, N, NHID1, NHID2, be);
  }

  // ---- layer 2: s = h @ W2 ; t = relu(spmm(s) + b2) (bf16) ----
  {
    dim3 grid(Mpad / 128, NHID2 / 64);
    k_gemm_bf16<<<grid, blk, 0, stream>>>(hbf, wt2, s_buf, N, NHID1, NHID2, nullptr);
  }
  (void)hipMemsetAsync(agg, 0, (size_t)N * NHID2 * 4, stream);
  {
    long long total = (long long)E * (NHID2 / 4);
    k_spmm_atomic<NHID2, NHID2 / 4><<<(unsigned)((total + 255) / 256), blk, 0, stream>>>(ei, ew, s_buf, agg, E);
  }
  k_bias_relu_bf16<NHID2><<<(unsigned)(((long long)N * NHID2 + 255) / 256), blk, 0, stream>>>(
      agg, b2, gbf, (long long)N * NHID2);
  (void)hipMemsetAsync(gbf + (size_t)N * NHID2, 0, (size_t)(Mpad - N) * NHID2 * 2, stream);

  // ---- layer 3: s = t @ W3 ; out1 = log_softmax(spmm(s) + b3) ----
  {
    dim3 grid(Mpad / 128, 1);
    k_gemm_bf16<<<grid, blk, 0, stream>>>(gbf, wt3, s_buf, N, NHID2, NCLASS, nullptr);
  }
  (void)hipMemsetAsync(agg, 0, (size_t)N * NCLASS * 4, stream);
  {
    long long total = (long long)E * (NCLASS / 4);
    k_spmm_atomic<NCLASS, NCLASS / 4><<<(unsigned)((total + 255) / 256), blk, 0, stream>>>(ei, ew, s_buf, agg, E);
  }
  k_logsoftmax40<<<(N + 7) / 8, blk, 0, stream>>>(agg, b3, out1, N);

  (void)n_in; (void)out_size; (void)ws_size; (void)off;
}
